// SCLinear_38173669327009
// MI455X (gfx1250) — 8-bit matrix operand order repaired and checked by reading, compiled, not run on hardware
//
#include <hip/hip_runtime.h>

typedef __attribute__((ext_vector_type(8))) int v8i;
typedef __attribute__((ext_vector_type(2))) short v2s;

#define TILE_M 64
#define TILE_N 32
#define KDIM   256

static __device__ __forceinline__ v2s as_v2s(unsigned u) {
    union { unsigned u; v2s s; } c; c.u = u; return c.s;
}

__device__ __forceinline__ float sanitize(float t) {
    t = (t != t) ? 0.0f : t;
    t = (fabsf(t) == __builtin_inff()) ? 1.0f : t;
    return t;
}

__device__ __forceinline__ float pow2_scale(float mx, float nf) {
    if (mx == 0.0f) mx = 1.0f;
    return exp2f(floorf(log2f(floorf(nf / mx))));
}

__global__ void sc_reduce_max(const float* __restrict__ x, int nx,
                              const float* __restrict__ W, int nw,
                              const float* __restrict__ b, int nb,
                              unsigned* __restrict__ maxes) {
    const int tid = blockIdx.x * blockDim.x + threadIdx.x;
    const int stride = gridDim.x * blockDim.x;
    float am = 0.0f, dm = 0.0f;
    for (int i = tid; i < nx; i += stride) am = fmaxf(am, fabsf(sanitize(x[i])));
    for (int i = tid; i < nw; i += stride) dm = fmaxf(dm, fabsf(sanitize(W[i])));
    for (int i = tid; i < nb; i += stride) dm = fmaxf(dm, fabsf(sanitize(b[i])));
    atomicMax(&maxes[0], __float_as_uint(am));
    atomicMax(&maxes[1], __float_as_uint(dm));
}

__global__ void __launch_bounds__(256)
sc_lut_gemm(const float* __restrict__ x, const float* __restrict__ W,
            const float* __restrict__ b, float* __restrict__ out,
            const unsigned* __restrict__ maxes,
            int M, int K, int O, int nshift) {
    __shared__ unsigned aaT[TILE_M * 65];
    __shared__ unsigned bbT[TILE_N * 65];

    const int tid = threadIdx.x;
    const int m0 = blockIdx.x * TILE_M;
    const int n0 = blockIdx.y * TILE_N;
    const float nf  = (float)(1 << nshift);
    const float sn2 = pow2_scale(__uint_as_float(maxes[0]), nf);
    const float sn1 = pow2_scale(__uint_as_float(maxes[1]), nf);

    for (int i = tid; i < TILE_M * 64; i += 256) {
        const int r = i >> 6, c4 = i & 63;
        const int gm = m0 + r;
        unsigned p = 0;
        #pragma unroll
        for (int j = 0; j < 4; ++j) {
            const float v = (gm < M) ? sanitize(x[gm * K + c4 * 4 + j]) : 0.0f;
            const int q = (int)(v * sn2);
            p |= ((unsigned)(q & 0xFF)) << (8 * j);
        }
        aaT[r * 65 + c4] = p;
    }
    for (int i = tid; i < TILE_N * 64; i += 256) {
        const int r = i >> 6, c4 = i & 63;
        const int gn = n0 + r;
        unsigned p = 0;
        #pragma unroll
        for (int j = 0; j < 4; ++j) {
            const float v = (gn < O) ? sanitize(W[gn * K + c4 * 4 + j]) : 0.0f;
            const int q = (int)(v * sn1);
            p |= ((unsigned)(q & 0xFF)) << (8 * j);
        }
        bbT[r * 65 + c4] = p;
    }
    __syncthreads();

    const int lane = tid & 31;
    const int wave = tid >> 5;
    const int wm = (wave & 3) * 16;
    const int wn = (wave >> 2) * 16;
    const int hi = lane >> 4;
    const int lm = lane & 15;

    v8i acc = {0, 0, 0, 0, 0, 0, 0, 0};
    #pragma unroll
    for (int kc = 0; kc < 4; ++kc) {
        v8i A, B;
        #pragma unroll
        for (int v = 0; v < 8; ++v) {
            const int ka = kc * 64 + (v >> 2) * 32 + hi * 16 + (v & 3) * 4;
            A[v] = (int)aaT[(wm + lm) * 65 + (ka >> 2)];
            const int kb = kc * 64 + (v >> 2) * 32 + hi * 16 + (v & 3) * 4;
            B[v] = (int)bbT[(wn + lm) * 65 + (kb >> 2)];
        }
        acc = __builtin_amdgcn_wmma_i32_16x16x64_iu8(
             false, A,  true, B, acc,
             false,  false);
    }

    const v2s S8  = {8, 8};
    const v2s S15 = {15, 15};
    const v2s CMK = {(short)((1 << nshift) - 1), (short)((1 << nshift) - 1)};
    v2s accp[8];
    #pragma unroll
    for (int g = 0; g < 8; ++g) accp[g] = (v2s){0, 0};
    v2s msum = {0, 0};
    const int nrow = wn + lm;
    #pragma unroll 4
    for (int c4 = 0; c4 < 64; ++c4) {
        const unsigned b4 = bbT[nrow * 65 + c4];
        const v2s blo = as_v2s((b4 << 8) & 0xFF00FF00u) >> S8;
        const v2s bhi = as_v2s(b4 & 0xFF00FF00u) >> S8;
        const v2s mlo = blo >> S15;
        const v2s mhi = bhi >> S15;
        const v2s balo = (blo ^ mlo) - mlo;
        const v2s bahi = (bhi ^ mhi) - mhi;
        msum = msum + mlo + mhi;
        #pragma unroll
        for (int g = 0; g < 8; ++g) {
            const unsigned a4 = aaT[(wm + g + hi * 8) * 65 + c4];
            const v2s alo = as_v2s(a4 & 0x00FF00FFu);
            const v2s ahi = as_v2s((a4 >> 8) & 0x00FF00FFu);
            const v2s rlo = (alo * balo) & CMK;
            const v2s rhi = (ahi * bahi) & CMK;
            accp[g] = accp[g] + ((rlo ^ mlo) + (rhi ^ mhi));
        }
    }
    const int msumTot = (int)msum.x + (int)msum.y;

    const int gn = n0 + wn + lm;
    const float bval = (gn < O) ? sanitize(b[gn]) : 0.0f;
    const int cc = (int)(bval * sn1);
    #pragma unroll
    for (int g = 0; g < 8; ++g) {
        const int gm = m0 + wm + g + hi * 8;
        if (gm < M && gn < O) {
            const int corr = (int)accp[g].x + (int)accp[g].y - msumTot;
            const int x9 = (acc[g] - corr) >> nshift;
            const int q  = (int)((float)x9 / sn2);
            out[gm * O + gn] = (float)(q + cc) / sn1;
        }
    }
}

extern "C" void kernel_launch(void* const* d_in, const int* in_sizes, int n_in,
                              void* d_out, int out_size, void* d_ws, size_t ws_size,
                              hipStream_t stream) {
    const float* x = (const float*)d_in[0];
    const float* W = (const float*)d_in[1];
    const float* b = (const float*)d_in[2];

    const int O = in_sizes[2];
    const int K = in_sizes[1] / O;
    const int M = in_sizes[0] / K;
    int root = 1;
    while ((root + 1) * (root + 1) <= in_sizes[3]) ++root;
    const int n = root - 1;
    int nshift = 0;
    while ((1 << (nshift + 1)) <= n) ++nshift;

    unsigned* maxes = (unsigned*)d_ws;
    hipMemsetAsync(d_ws, 0, 2 * sizeof(unsigned), stream);

    sc_reduce_max<<<64, 256, 0, stream>>>(x, M * K, W, O * K, b, O, maxes);

    dim3 grid((M + TILE_M - 1) / TILE_M, (O + TILE_N - 1) / TILE_N);
    sc_lut_gemm<<<grid, 256, 0, stream>>>(x, W, b, (float*)d_out, maxes, M, K, O, nshift);
}
